// GmpnnBlock_4269197492826
// MI455X (gfx1250) — matrix operand order repaired and checked by reading, not compiled, not run on hardware
//
#include <hip/hip_runtime.h>
#include <hip/hip_bf16.h>
#include <math.h>

// ---------------------------------------------------------------------------
// CDNA5 / gfx1250 WMMA types & helpers (wave32, v_wmma_f32_16x16x32_bf16)
// ---------------------------------------------------------------------------
typedef __attribute__((ext_vector_type(16))) __bf16 bfrag;   // 8 VGPRs
typedef __attribute__((ext_vector_type(8)))  float  v8f;     // 8 VGPRs

__device__ __forceinline__ v8f wmma_bf(bfrag a, bfrag b, v8f c) {
  // (neg_a, A, neg_b, B, c_mod, C, reuse_a, reuse_b)
  return __builtin_amdgcn_wmma_f32_16x16x32_bf16(false, a, false, b,
                                                 (short)0, c, false, false);
}

// ---------------------------------------------------------------------------
// LDS weight staging, pre-swizzled into WMMA B-fragment order:
//   sW[frag*512 + lane*16 + e]  with  frag = kf*(NC/16) + nb
//   B (32x16) layout: lanes 0..15 -> K=kb+0..7,kb+16..23 at col n=nb*16+lane,
//                     lanes 16..31 -> K=kb+8..15,kb+24..31.
// Fragment fetch is then one contiguous 32B chunk per lane (2x ds_load_b128).
// ---------------------------------------------------------------------------
template<int K, int NC>
__device__ __forceinline__ void stage_w_swizzled(const float* __restrict__ W,
                                                 __bf16* sW) {
  constexpr int NB = NC / 16;
  for (int i = threadIdx.x; i < K * NC; i += 256) {
    int e    = i & 15;
    int lane = (i >> 4) & 31;
    int frag = i >> 9;                 // /512
    int kf = frag / NB, nb = frag - kf * NB;
    int n  = nb * 16 + (lane & 15);
    int kk = kf * 32 + ((e & 8) ? 16 : 0) + ((lane & 16) ? 8 : 0) + (e & 7);
    sW[i] = (__bf16)W[kk * NC + n];
  }
}

__device__ __forceinline__ bfrag load_b_frag_sw(const __bf16* sW, int frag,
                                                int lane) {
  return *(const bfrag*)(sW + frag * 512 + lane * 16);
}

// Pack a transformed float into fragment slot
#define PRELU(v, s) ((v) >= 0.0f ? (v) : (s) * (v))

// ---------------------------------------------------------------------------
// Generic WMMA GEMM:  Y = act(X*scale + shift) @ W + bb ; optional (Y+R)*0.5
// X:[Nrows,K] f32 row-major, W:[K,NC] f32 row-major (staged bf16 in LDS).
// One wave computes a 16-row strip; 8 waves / 256-thread block = 128 rows.
// ---------------------------------------------------------------------------
template<int K, int NC>
__global__ __launch_bounds__(256)
void k_gemm(const float* __restrict__ X, int Nrows,
            const float* __restrict__ scale, const float* __restrict__ shift,
            const float* __restrict__ prelu_p,
            const float* __restrict__ W, const float* __restrict__ bb,
            const float* __restrict__ R, float* __restrict__ Y) {
  __shared__ __attribute__((aligned(32))) __bf16 sW[K * NC];
  __shared__ __attribute__((aligned(16))) float sScale[K];
  __shared__ __attribute__((aligned(16))) float sShift[K];
  __shared__ float sBb[NC];
  stage_w_swizzled<K, NC>(W, sW);
  for (int i = threadIdx.x; i < K; i += 256) {
    sScale[i] = scale ? scale[i] : 1.0f;
    sShift[i] = shift ? shift[i] : 0.0f;
  }
  for (int i = threadIdx.x; i < NC; i += 256) sBb[i] = bb ? bb[i] : 0.0f;
  __syncthreads();

  const float slope = prelu_p ? prelu_p[0] : 1.0f;
  const int wave = threadIdx.x >> 5, lane = threadIdx.x & 31;
  const int rowBase = blockIdx.x * 128 + wave * 16;
  const int row  = rowBase + (lane & 15);
  const int rowc = (row < Nrows) ? row : (Nrows - 1);
  const float4* x4  = (const float4*)(X + (size_t)rowc * K);
  const float4* sc4 = (const float4*)sScale;
  const float4* sh4 = (const float4*)sShift;
  const int kh = (lane & 16) ? 8 : 0;

  constexpr int KF = K / 32;
  constexpr int NB = NC / 16;
  bfrag afr[KF];
#pragma unroll
  for (int kf = 0; kf < KF; ++kf) {
    // elements 0..7  <- K = kb+kh   .. kb+kh+7   (two float4)
    // elements 8..15 <- K = kb+16+kh.. kb+16+kh+7 (two float4)
    const int q0 = (kf * 32 + kh) >> 2;        // /4, multiple of 2
    const int q1 = (kf * 32 + 16 + kh) >> 2;
    float4 a0 = x4[q0], a1 = x4[q0 + 1], a2 = x4[q1], a3 = x4[q1 + 1];
    float4 s0 = sc4[q0], s1 = sc4[q0 + 1], s2 = sc4[q1], s3 = sc4[q1 + 1];
    float4 t0 = sh4[q0], t1 = sh4[q0 + 1], t2 = sh4[q1], t3 = sh4[q1 + 1];
    bfrag f;
    f[0]  = (__bf16)PRELU(a0.x * s0.x + t0.x, slope);
    f[1]  = (__bf16)PRELU(a0.y * s0.y + t0.y, slope);
    f[2]  = (__bf16)PRELU(a0.z * s0.z + t0.z, slope);
    f[3]  = (__bf16)PRELU(a0.w * s0.w + t0.w, slope);
    f[4]  = (__bf16)PRELU(a1.x * s1.x + t1.x, slope);
    f[5]  = (__bf16)PRELU(a1.y * s1.y + t1.y, slope);
    f[6]  = (__bf16)PRELU(a1.z * s1.z + t1.z, slope);
    f[7]  = (__bf16)PRELU(a1.w * s1.w + t1.w, slope);
    f[8]  = (__bf16)PRELU(a2.x * s2.x + t2.x, slope);
    f[9]  = (__bf16)PRELU(a2.y * s2.y + t2.y, slope);
    f[10] = (__bf16)PRELU(a2.z * s2.z + t2.z, slope);
    f[11] = (__bf16)PRELU(a2.w * s2.w + t2.w, slope);
    f[12] = (__bf16)PRELU(a3.x * s3.x + t3.x, slope);
    f[13] = (__bf16)PRELU(a3.y * s3.y + t3.y, slope);
    f[14] = (__bf16)PRELU(a3.z * s3.z + t3.z, slope);
    f[15] = (__bf16)PRELU(a3.w * s3.w + t3.w, slope);
    afr[kf] = f;
  }

  const int rbase = (lane & 16) ? 8 : 0;
#pragma unroll
  for (int nb = 0; nb < NB; ++nb) {
    v8f acc = {};
#pragma unroll
    for (int kf = 0; kf < KF; ++kf)
      acc = wmma_bf(afr[kf], load_b_frag_sw(sW, kf * NB + nb, lane), acc);
    const int col = nb * 16 + (lane & 15);
    const float bbv = sBb[col];
#pragma unroll
    for (int r = 0; r < 8; ++r) {
      int gr = rowBase + rbase + r;
      if (gr < Nrows) {
        float v = acc[r] + bbv;
        if (R) v = (v + R[(size_t)gr * NC + col]) * 0.5f;
        Y[(size_t)gr * NC + col] = v;
      }
    }
  }
}

// ---------------------------------------------------------------------------
// Fused edge-weight kernel:
//   a = prelu(ax[dst]+ay[src]+bias, p) @ W_sml + b_sml      (WMMA)
//   ew = sigmoid( dot(a, ef) * invdeg[src] )                (shfl reduce)
// ---------------------------------------------------------------------------
__global__ __launch_bounds__(256)
void k_edge_ew(const int* __restrict__ src, const int* __restrict__ dst,
               const float* __restrict__ ax, const float* __restrict__ ay,
               const float* __restrict__ bias, const float* __restrict__ p_sml,
               const float* __restrict__ Wsml, const float* __restrict__ bsml,
               const float* __restrict__ ef, const float* __restrict__ invdeg,
               float* __restrict__ ew, int E) {
  __shared__ __attribute__((aligned(32))) __bf16 sW[64 * 64];
  __shared__ __attribute__((aligned(16))) float sBias[64];
  __shared__ float sBs[64];
  stage_w_swizzled<64, 64>(Wsml, sW);
  for (int i = threadIdx.x; i < 64; i += 256) { sBias[i] = bias[i]; sBs[i] = bsml[i]; }
  __syncthreads();

  const float p = p_sml[0];
  const int wave = threadIdx.x >> 5, lane = threadIdx.x & 31;
  const int rowBase = blockIdx.x * 128 + wave * 16;
  const int m = lane & 15;
  int e = rowBase + m; if (e >= E) e = E - 1;
  const int s = src[e], d = dst[e];
  const float4* ax4 = (const float4*)(ax + (size_t)d * 64);
  const float4* ay4 = (const float4*)(ay + (size_t)s * 64);
  const float4* bi4 = (const float4*)sBias;
  const int kh = (lane & 16) ? 8 : 0;

  bfrag afr[2];
#pragma unroll
  for (int kf = 0; kf < 2; ++kf) {
    const int q0 = (kf * 32 + kh) >> 2;
    const int q1 = (kf * 32 + 16 + kh) >> 2;
    float4 u0 = ax4[q0], u1 = ax4[q0 + 1], u2 = ax4[q1], u3 = ax4[q1 + 1];
    float4 v0 = ay4[q0], v1 = ay4[q0 + 1], v2 = ay4[q1], v3 = ay4[q1 + 1];
    float4 c0 = bi4[q0], c1 = bi4[q0 + 1], c2 = bi4[q1], c3 = bi4[q1 + 1];
    bfrag f;
    f[0]  = (__bf16)PRELU(u0.x + v0.x + c0.x, p);
    f[1]  = (__bf16)PRELU(u0.y + v0.y + c0.y, p);
    f[2]  = (__bf16)PRELU(u0.z + v0.z + c0.z, p);
    f[3]  = (__bf16)PRELU(u0.w + v0.w + c0.w, p);
    f[4]  = (__bf16)PRELU(u1.x + v1.x + c1.x, p);
    f[5]  = (__bf16)PRELU(u1.y + v1.y + c1.y, p);
    f[6]  = (__bf16)PRELU(u1.z + v1.z + c1.z, p);
    f[7]  = (__bf16)PRELU(u1.w + v1.w + c1.w, p);
    f[8]  = (__bf16)PRELU(u2.x + v2.x + c2.x, p);
    f[9]  = (__bf16)PRELU(u2.y + v2.y + c2.y, p);
    f[10] = (__bf16)PRELU(u2.z + v2.z + c2.z, p);
    f[11] = (__bf16)PRELU(u2.w + v2.w + c2.w, p);
    f[12] = (__bf16)PRELU(u3.x + v3.x + c3.x, p);
    f[13] = (__bf16)PRELU(u3.y + v3.y + c3.y, p);
    f[14] = (__bf16)PRELU(u3.z + v3.z + c3.z, p);
    f[15] = (__bf16)PRELU(u3.w + v3.w + c3.w, p);
    afr[kf] = f;
  }

  const int rbase = (lane & 16) ? 8 : 0;
  float part[8] = {0.f, 0.f, 0.f, 0.f, 0.f, 0.f, 0.f, 0.f};
#pragma unroll
  for (int nb = 0; nb < 4; ++nb) {
    v8f acc = {};
    acc = wmma_bf(afr[0], load_b_frag_sw(sW, 0 * 4 + nb, lane), acc);
    acc = wmma_bf(afr[1], load_b_frag_sw(sW, 1 * 4 + nb, lane), acc);
    const int col = nb * 16 + m;
    const float bsv = sBs[col];
#pragma unroll
    for (int r = 0; r < 8; ++r) {
      int ge = rowBase + rbase + r; if (ge >= E) ge = E - 1;
      part[r] += (acc[r] + bsv) * ef[(size_t)ge * 64 + col];
    }
  }
#pragma unroll
  for (int r = 0; r < 8; ++r) {
#pragma unroll
    for (int mask = 1; mask <= 8; mask <<= 1)
      part[r] += __shfl_xor(part[r], mask, 32);
  }
  if (m == 0) {
#pragma unroll
    for (int r = 0; r < 8; ++r) {
      int ge = rowBase + rbase + r;
      if (ge < E) {
        float a = part[r] * invdeg[src[ge]];
        ew[ge] = 1.0f / (1.0f + __expf(-a));
      }
    }
  }
}

// ---------------------------------------------------------------------------
// Graph / elementwise kernels
// ---------------------------------------------------------------------------
__global__ void k_deg(const int* __restrict__ dst, float* __restrict__ deg, int E) {
  int e = blockIdx.x * blockDim.x + threadIdx.x;
  if (e < E) atomicAdd(&deg[dst[e]], 1.0f);
}
__global__ void k_inv(float* __restrict__ deg, int N) {
  int i = blockIdx.x * blockDim.x + threadIdx.x;
  if (i < N) deg[i] = 1.0f / deg[i];
}
// out = (x[src] + msg) * ew   (msg==nullptr -> out = x[src]*ew)
__global__ void k_combine(const float* __restrict__ x, const int* __restrict__ src,
                          const float* __restrict__ ew, const float* __restrict__ msg,
                          float* __restrict__ out, int E) {
  int i = blockIdx.x * blockDim.x + threadIdx.x;
  if (i >= E * 16) return;
  int e = i >> 4, c = (i & 15) << 2;
  float w = ew[e]; int s = src[e];
  float4 xv = *(const float4*)(x + (size_t)s * 64 + c);
  float4 o;
  if (msg) {
    float4 mv = *(const float4*)(msg + (size_t)e * 64 + c);
    o.x = (xv.x + mv.x) * w; o.y = (xv.y + mv.y) * w;
    o.z = (xv.z + mv.z) * w; o.w = (xv.w + mv.w) * w;
  } else {
    o.x = xv.x * w; o.y = xv.y * w; o.z = xv.z * w; o.w = xv.w * w;
  }
  *(float4*)(out + (size_t)e * 64 + c) = o;
}
__global__ void k_scatter_lg(const int* __restrict__ lgs, const int* __restrict__ lgd,
                             const float* __restrict__ out, float* __restrict__ msg, int L) {
  int i = blockIdx.x * blockDim.x + threadIdx.x;
  if (i >= L * 16) return;
  int l = i >> 4, c = (i & 15) << 2;
  int es = lgs[l], ed = lgd[l];
  float4 v = *(const float4*)(out + (size_t)es * 64 + c);
  float* mp = msg + (size_t)ed * 64 + c;
  atomicAdd(mp + 0, v.x); atomicAdd(mp + 1, v.y);
  atomicAdd(mp + 2, v.z); atomicAdd(mp + 3, v.w);
}
__global__ void k_scatter_h(const int* __restrict__ dst, const float* __restrict__ out,
                            float* __restrict__ h, int E) {
  int i = blockIdx.x * blockDim.x + threadIdx.x;
  if (i >= E * 16) return;
  int e = i >> 4, c = (i & 15) << 2;
  int dn = dst[e];
  float4 v = *(const float4*)(out + (size_t)e * 64 + c);
  float* hp = h + (size_t)dn * 64 + c;
  atomicAdd(hp + 0, v.x); atomicAdd(hp + 1, v.y);
  atomicAdd(hp + 2, v.z); atomicAdd(hp + 3, v.w);
}
// BatchNorm column stats (biased) and affine fold
__global__ void k_colstats(const float* __restrict__ A, int Nrows, int C,
                           float* __restrict__ sum, float* __restrict__ sumsq) {
  int col = threadIdx.x; if (col >= C) return;
  float s = 0.f, s2 = 0.f;
  for (int r = blockIdx.x; r < Nrows; r += gridDim.x) {
    float v = A[(size_t)r * C + col]; s += v; s2 += v * v;
  }
  atomicAdd(&sum[col], s); atomicAdd(&sumsq[col], s2);
}
__global__ void k_bnfinal(const float* __restrict__ sum, const float* __restrict__ sumsq,
                          const float* __restrict__ g, const float* __restrict__ b,
                          int C, float Ninv, float* __restrict__ sc, float* __restrict__ sh) {
  int c = threadIdx.x; if (c >= C) return;
  float m = sum[c] * Ninv;
  float var = sumsq[c] * Ninv - m * m;
  float s = g[c] * rsqrtf(var + 1e-5f);
  sc[c] = s; sh[c] = b[c] - m * s;
}

// ---------------------------------------------------------------------------
// Launch sequence
// ---------------------------------------------------------------------------
extern "C" void kernel_launch(void* const* d_in, const int* in_sizes, int n_in,
                              void* d_out, int out_size, void* d_ws, size_t ws_size,
                              hipStream_t stream) {
  const float* x    = (const float*)d_in[0];
  const float* efr  = (const float*)d_in[1];
  const int*   ei   = (const int*)d_in[2];
  const int*   lg   = (const int*)d_in[3];
  const float* w_i  = (const float*)d_in[4];
  const float* w_j  = (const float*)d_in[5];
  const float* bias = (const float*)d_in[6];
  const float* We   = (const float*)d_in[7];
  const float* be   = (const float*)d_in[8];
  const float* psml = (const float*)d_in[9];
  const float* Wsml = (const float*)d_in[10];
  const float* bsml = (const float*)d_in[11];
  const float* g1 = (const float*)d_in[12]; const float* b1 = (const float*)d_in[13];
  const float* W1 = (const float*)d_in[14]; const float* bb1 = (const float*)d_in[15];
  const float* g2 = (const float*)d_in[16]; const float* b2 = (const float*)d_in[17];
  const float* p2 = (const float*)d_in[18]; const float* W2 = (const float*)d_in[19];
  const float* bb2 = (const float*)d_in[20];
  const float* g3 = (const float*)d_in[21]; const float* b3 = (const float*)d_in[22];
  const float* p3 = (const float*)d_in[23]; const float* W3 = (const float*)d_in[24];
  const float* bb3 = (const float*)d_in[25];
  const float* g4 = (const float*)d_in[26]; const float* b4 = (const float*)d_in[27];
  const float* p4 = (const float*)d_in[28]; const float* W4 = (const float*)d_in[29];
  const float* bb4 = (const float*)d_in[30];

  const int N = in_sizes[0] / 64;
  const int E = in_sizes[2] / 2;
  const int L = in_sizes[3] / 2;
  const int* src = ei;       const int* dst = ei + E;
  const int* lgs = lg;       const int* lgd = lg + L;

  // workspace carve-out (all offsets multiples of 4 floats -> float4 safe)
  float* ws = (float*)d_ws; size_t o = 0;
  float* ef   = ws + o; o += (size_t)E * 64;   // reused as msg after k_edge_ew
  float* outb = ws + o; o += (size_t)E * 64;
  float* ax   = ws + o; o += (size_t)N * 64;
  float* ay   = ws + o; o += (size_t)N * 64;
  float* idg  = ws + o; o += (size_t)N;
  float* ewp  = ws + o; o += (size_t)E;
  float* hbuf = ws + o; o += (size_t)N * 64;
  float* Hb1  = ws + o; o += (size_t)N * 128;
  float* Hb2  = ws + o; o += (size_t)N * 128;
  float* Hb3  = ws + o; o += (size_t)N * 128;
  float* ssum = ws + o; o += 128;
  float* ssq  = ws + o; o += 128;
  float* scb  = ws + o; o += 128;
  float* shb  = ws + o; o += 128;
  float* msg  = ef;

  const float Ninv = 1.0f / (float)N;
  const int gE16 = (E * 16 + 255) / 256;
  const int gL16 = (L * 16 + 255) / 256;

  // 1) degree + inverse
  hipMemsetAsync(idg, 0, (size_t)N * 4, stream);
  k_deg<<<(E + 255) / 256, 256, 0, stream>>>(dst, idg, E);
  k_inv<<<(N + 255) / 256, 256, 0, stream>>>(idg, N);

  // 2) ef = edge_feats_raw @ We + be   [E,32]x[32,64]
  k_gemm<32, 64><<<(E + 127) / 128, 256, 0, stream>>>(
      efr, E, nullptr, nullptr, nullptr, We, be, nullptr, ef);

  // 3) ax = x @ w_i ; ay = x @ w_j    [N,64]x[64,64]
  k_gemm<64, 64><<<(N + 127) / 128, 256, 0, stream>>>(
      x, N, nullptr, nullptr, nullptr, w_i, nullptr, nullptr, ax);
  k_gemm<64, 64><<<(N + 127) / 128, 256, 0, stream>>>(
      x, N, nullptr, nullptr, nullptr, w_j, nullptr, nullptr, ay);

  // 4) edge weights (fused WMMA + dot + sigmoid)
  k_edge_ew<<<(E + 127) / 128, 256, 0, stream>>>(
      src, dst, ax, ay, bias, psml, Wsml, bsml, ef, idg, ewp, E);

  // 5) out = edge_attr = x[src]*ew
  k_combine<<<gE16, 256, 0, stream>>>(x, src, ewp, nullptr, outb, E);

  // 6) three message-passing iterations
  for (int it = 0; it < 3; ++it) {
    hipMemsetAsync(msg, 0, (size_t)E * 64 * 4, stream);
    k_scatter_lg<<<gL16, 256, 0, stream>>>(lgs, lgd, outb, msg, L);
    k_combine<<<gE16, 256, 0, stream>>>(x, src, ewp, msg, outb, E);
  }

  // 7) h = x + segment_sum(out, dst)
  hipMemcpyAsync(hbuf, x, (size_t)N * 64 * 4, hipMemcpyDeviceToDevice, stream);
  k_scatter_h<<<gE16, 256, 0, stream>>>(dst, outb, hbuf, E);

  // 8) MLP: H = bn(h) @ W1 + bb1
  hipMemsetAsync(ssum, 0, 256 * 4, stream);
  k_colstats<<<256, 64, 0, stream>>>(hbuf, N, 64, ssum, ssq);
  k_bnfinal<<<1, 64, 0, stream>>>(ssum, ssq, g1, b1, 64, Ninv, scb, shb);
  k_gemm<64, 128><<<(N + 127) / 128, 256, 0, stream>>>(
      hbuf, N, scb, shb, nullptr, W1, bb1, nullptr, Hb1);

  // 9) h2 = prelu(bn(H),p2) @ W2 + bb2
  hipMemsetAsync(ssum, 0, 256 * 4, stream);
  k_colstats<<<256, 128, 0, stream>>>(Hb1, N, 128, ssum, ssq);
  k_bnfinal<<<1, 128, 0, stream>>>(ssum, ssq, g2, b2, 128, Ninv, scb, shb);
  k_gemm<128, 128><<<(N + 127) / 128, 256, 0, stream>>>(
      Hb1, N, scb, shb, p2, W2, bb2, nullptr, Hb2);

  // 10) Hb = (prelu(bn(h2),p3) @ W3 + bb3 + H) / 2
  hipMemsetAsync(ssum, 0, 256 * 4, stream);
  k_colstats<<<256, 128, 0, stream>>>(Hb2, N, 128, ssum, ssq);
  k_bnfinal<<<1, 128, 0, stream>>>(ssum, ssq, g3, b3, 128, Ninv, scb, shb);
  k_gemm<128, 128><<<(N + 127) / 128, 256, 0, stream>>>(
      Hb2, N, scb, shb, p3, W3, bb3, Hb1, Hb3);

  // 11) out = (prelu(bn(Hb),p4) @ W4 + bb4 + Hb) / 2  -> d_out
  hipMemsetAsync(ssum, 0, 256 * 4, stream);
  k_colstats<<<256, 128, 0, stream>>>(Hb3, N, 128, ssum, ssq);
  k_bnfinal<<<1, 128, 0, stream>>>(ssum, ssq, g4, b4, 128, Ninv, scb, shb);
  k_gemm<128, 128><<<(N + 127) / 128, 256, 0, stream>>>(
      Hb3, N, scb, shb, p4, W4, bb4, Hb3, (float*)d_out);
}
